// GraphAttentionModel_44727789421270
// MI455X (gfx1250) — matrix operand order repaired and checked by reading, not compiled, not run on hardware
//
#include <hip/hip_runtime.h>

// ---------------------------------------------------------------------------
// Problem constants (from reference)
// ---------------------------------------------------------------------------
#define G_   16
#define NPG_ 2048
#define KNN_ 16
#define H_   8
#define F_   64
#define L_   4
#define DM_  512            // H*F
#define NN_  (G_ * NPG_)    // 32768
#define DIN_ 6
#define DOUT_ 16

typedef __attribute__((ext_vector_type(16))) __bf16 v16bf;
typedef __attribute__((ext_vector_type(8)))  __bf16 v8bf;
typedef __attribute__((ext_vector_type(8)))  float  v8f;

__device__ __forceinline__ __bf16 f2bf(float f) {
    unsigned u = __builtin_bit_cast(unsigned, f);
    unsigned r = u + 0x7FFFu + ((u >> 16) & 1u);   // round-to-nearest-even
    unsigned short s = (unsigned short)(r >> 16);
    return __builtin_bit_cast(__bf16, s);
}

// LDS byte offset of a shared-memory pointer (addrspace(3) is 32-bit).
__device__ __forceinline__ unsigned lds_off(const void* p) {
    return (unsigned)(uintptr_t)(__attribute__((address_space(3))) const void*)p;
}

// Async global->LDS 16B copy (per lane), GVS addressing: saddr + 32-bit voffset.
// Tracked by ASYNCcnt.
__device__ __forceinline__ void async_copy_b128(unsigned lds, const void* gbase,
                                                unsigned goff_bytes) {
    asm volatile("global_load_async_to_lds_b128 %0, %1, %2"
                 :: "v"(lds), "v"(goff_bytes), "s"(gbase)
                 : "memory");
}

// ---------------------------------------------------------------------------
// KNN: per-graph brute-force top-16 smallest d2 (self excluded).
// One thread per target node; candidates staged through LDS.
// ---------------------------------------------------------------------------
__global__ void knn_kernel(const float* __restrict__ pos, int* __restrict__ srcIdx) {
    const int t     = blockIdx.x * 256 + threadIdx.x;
    const int gph   = t >> 11;          // /NPG_
    const int gbase = gph << 11;

    const float px = pos[t * 3 + 0];
    const float py = pos[t * 3 + 1];
    const float pz = pos[t * 3 + 2];

    float dist[KNN_];
    int   idx[KNN_];
#pragma unroll
    for (int j = 0; j < KNN_; ++j) { dist[j] = 1e30f; idx[j] = gbase; }

    __shared__ float sx[256], sy[256], sz[256];

    for (int ch = 0; ch < NPG_ / 256; ++ch) {
        const int c = gbase + ch * 256 + threadIdx.x;
        sx[threadIdx.x] = pos[c * 3 + 0];
        sy[threadIdx.x] = pos[c * 3 + 1];
        sz[threadIdx.x] = pos[c * 3 + 2];
        __syncthreads();
        const int cbase = gbase + ch * 256;
        for (int i = 0; i < 256; ++i) {
            const int cand = cbase + i;
            if (cand == t) continue;
            const float dx = px - sx[i], dy = py - sy[i], dz = pz - sz[i];
            const float d2 = dx * dx + dy * dy + dz * dz;
            if (d2 < dist[KNN_ - 1]) {
                float nd = d2; int ni = cand;
#pragma unroll
                for (int j = 0; j < KNN_; ++j) {        // sorted insert (reg-resident)
                    const bool sw = nd < dist[j];
                    const float td = dist[j]; const int ti = idx[j];
                    dist[j] = sw ? nd : td;  idx[j] = sw ? ni : ti;
                    nd      = sw ? td : nd;  ni     = sw ? ti : ni;
                }
            }
        }
        __syncthreads();
    }
#pragma unroll
    for (int j = 0; j < KNN_; ++j) srcIdx[t * KNN_ + j] = idx[j];
}

// ---------------------------------------------------------------------------
// Encoder stage 1: h1 = relu(cat(x,pos) @ W1 + b1)  -> bf16 [NN, 256]
// ---------------------------------------------------------------------------
__global__ void enc1_kernel(const float* __restrict__ x, const float* __restrict__ pos,
                            const float* __restrict__ W1, const float* __restrict__ b1,
                            __bf16* __restrict__ h1) {
    const int n = blockIdx.x;
    const int o = threadIdx.x;            // 256
    float s = b1[o];
#pragma unroll
    for (int d = 0; d < 3; ++d) s += x[n * 3 + d] * W1[d * 256 + o];
#pragma unroll
    for (int d = 0; d < 3; ++d) s += pos[n * 3 + d] * W1[(3 + d) * 256 + o];
    h1[(size_t)n * 256 + o] = f2bf(fmaxf(s, 0.f));
}

// ---------------------------------------------------------------------------
// Weight convert + transpose: f32 [K][N] row-major -> bf16 [N][K]
// ---------------------------------------------------------------------------
__global__ void convt_kernel(const float* __restrict__ W, __bf16* __restrict__ WT,
                             int Kd, int N) {
    const int i = blockIdx.x * 256 + threadIdx.x;
    if (i >= Kd * N) return;
    const int k = i / N, n = i - k * N;
    WT[(size_t)n * Kd + k] = f2bf(W[i]);
}

// ---------------------------------------------------------------------------
// bf16 WMMA GEMM: C[M,N] = A[M,K] @ BT[N,K]^T (+bias) (+relu)
//
// Block = 256 threads (8 waves), block tile 64(M) x 128(N).
// Waves arranged 4(M) x 2(N); each wave computes 16x64 via 4 WMMA accumulators,
// reusing its A fragment 4x. K consumed in chunks of 32.
// A panel (64x32) and B panel (128x32) double-buffered in LDS, filled with
// async global->LDS b128 copies (ASYNCcnt) overlapped with WMMA compute.
// ---------------------------------------------------------------------------
__global__ void __launch_bounds__(256)
gemm_bf16_wmma(const __bf16* __restrict__ A, const __bf16* __restrict__ BT,
               const float* __restrict__ bias,
               float* __restrict__ Cf, __bf16* __restrict__ Cb,
               int M, int N, int Kd, int relu) {
    __shared__ __bf16 As[2][64 * 32];    // [row][k] row-major, 4KB per buffer
    __shared__ __bf16 Bs[2][128 * 32];   // [col][k] row-major, 8KB per buffer

    const int tid  = threadIdx.x;
    const int lane = tid & 31;
    const int wave = tid >> 5;

    const int bnCnt = N >> 7;
    const int bm = blockIdx.x / bnCnt;
    const int bn = blockIdx.x - bm * bnCnt;
    const int m0 = bm << 6;
    const int n0 = bn << 7;

    const int wm = wave >> 1;            // 0..3  (16-row strip)
    const int wn = wave & 1;             // 0..1  (64-col strip)
    const int l15 = lane & 15;
    const int hi  = lane >> 4;           // 0 or 1
    const int akh = hi << 3;             // A k-offset: 0 or 8
    const int bkh = hi << 3;             // B k-offset: 0 or 8

    // staging split: A = 256 x 16B, B = 512 x 16B
    const int arow = tid >> 2, aseg = tid & 3;

    v8f acc[4];
#pragma unroll
    for (int nf = 0; nf < 4; ++nf) {
        const float bv = bias ? bias[n0 + (wn << 6) + (nf << 4) + l15] : 0.f;
#pragma unroll
        for (int j = 0; j < 8; ++j) acc[nf][j] = bv;
    }

    const unsigned aldsBase[2] = { lds_off(&As[0][0]), lds_off(&As[1][0]) };
    const unsigned bldsBase[2] = { lds_off(&Bs[0][0]), lds_off(&Bs[1][0]) };

    auto issue = [&](int kc, int buf) {
        const int k0 = kc << 5;
        // A panel: one 16B copy per thread
        {
            const unsigned lds = aldsBase[buf] + (unsigned)(arow * 64 + aseg * 16);
            const unsigned go  = (unsigned)((((size_t)(m0 + arow)) * Kd + k0 + (aseg << 3)) << 1);
            async_copy_b128(lds, A, go);
        }
        // B panel: two 16B copies per thread
#pragma unroll
        for (int j = 0; j < 2; ++j) {
            const int s  = (tid << 1) + j;
            const int br = s >> 2, bsg = s & 3;
            const unsigned lds = bldsBase[buf] + (unsigned)(br * 64 + bsg * 16);
            const unsigned go  = (unsigned)((((size_t)(n0 + br)) * Kd + k0 + (bsg << 3)) << 1);
            async_copy_b128(lds, BT, go);
        }
    };

    const int nk = Kd >> 5;
    issue(0, 0);

    for (int kc = 0; kc < nk; ++kc) {
        const int buf = kc & 1;
        if (kc + 1 < nk) {
            issue(kc + 1, buf ^ 1);                         // 3 more in flight
            asm volatile("s_wait_asynccnt 3" ::: "memory"); // chunk kc landed
        } else {
            asm volatile("s_wait_asynccnt 0" ::: "memory");
        }
        __syncthreads();

        const __bf16* __restrict__ Ab = &As[buf][0];
        const __bf16* __restrict__ Bb = &Bs[buf][0];

        const v8bf alo = *(const v8bf*)(Ab + ((wm << 4) + l15) * 32 + akh);
        const v8bf ahi = *(const v8bf*)(Ab + ((wm << 4) + l15) * 32 + 16 + akh);
        v16bf a;
#pragma unroll
        for (int i = 0; i < 8; ++i) { a[i] = alo[i]; a[8 + i] = ahi[i]; }

#pragma unroll
        for (int nf = 0; nf < 4; ++nf) {
            const int col = (wn << 6) + (nf << 4) + l15;
            const v8bf blo = *(const v8bf*)(Bb + col * 32 + bkh);
            const v8bf bhi = *(const v8bf*)(Bb + col * 32 + 16 + bkh);
            v16bf b;
#pragma unroll
            for (int i = 0; i < 8; ++i) { b[i] = blo[i]; b[8 + i] = bhi[i]; }
            acc[nf] = __builtin_amdgcn_wmma_f32_16x16x32_bf16(
                          false, a, false, b, (short)0, acc[nf], false, false);
        }
        __syncthreads();
    }

    const int rbase = m0 + (wm << 4) + (hi << 3);
#pragma unroll
    for (int nf = 0; nf < 4; ++nf) {
        const int col = n0 + (wn << 6) + (nf << 4) + l15;
#pragma unroll
        for (int j = 0; j < 8; ++j) {
            float v = acc[nf][j];
            if (relu) v = fmaxf(v, 0.f);
            const size_t off = (size_t)(rbase + j) * N + col;
            if (Cf) Cf[off] = v;
            if (Cb) Cb[off] = f2bf(v);
        }
    }
}

// ---------------------------------------------------------------------------
// Residual + LayerNorm: out = LN(a [+ res]) * g + b ; optional f32/bf16 outs.
// blockDim = 512, one block per node.
// ---------------------------------------------------------------------------
__global__ void ln_kernel(const float* __restrict__ a, const float* __restrict__ res,
                          const float* __restrict__ g, const float* __restrict__ b,
                          float* __restrict__ outF, __bf16* __restrict__ outB) {
    const int n = blockIdx.x;
    const int d = threadIdx.x;
    __shared__ float s1[DM_], s2[DM_];
    float v = a[(size_t)n * DM_ + d];
    if (res) v += res[(size_t)n * DM_ + d];
    s1[d] = v; s2[d] = v * v;
    __syncthreads();
    for (int s = DM_ / 2; s > 0; s >>= 1) {
        if (d < s) { s1[d] += s1[d + s]; s2[d] += s2[d + s]; }
        __syncthreads();
    }
    const float mu  = s1[0] * (1.f / DM_);
    const float var = s2[0] * (1.f / DM_) - mu * mu;
    const float o = (v - mu) * rsqrtf(var + 1e-5f) * g[d] + b[d];
    if (outF) outF[(size_t)n * DM_ + d] = o;
    if (outB) outB[(size_t)n * DM_ + d] = f2bf(o);
}

// ---------------------------------------------------------------------------
// Attention scores: es/ed [NN, H] from projected features hp [NN, 512]
// ---------------------------------------------------------------------------
__global__ void att_score_kernel(const float* __restrict__ hp,
                                 const float* __restrict__ aS, const float* __restrict__ aD,
                                 float* __restrict__ es, float* __restrict__ ed) {
    const int i = blockIdx.x * 256 + threadIdx.x;      // n*H + h
    const int hh = i & (H_ - 1);
    const int n  = i >> 3;
    const float* __restrict__ row = hp + (size_t)n * DM_ + hh * F_;
    const float* __restrict__ as  = aS + hh * F_;
    const float* __restrict__ ad  = aD + hh * F_;
    float s = 0.f, d = 0.f;
#pragma unroll 8
    for (int f = 0; f < F_; ++f) { s += row[f] * as[f]; d += row[f] * ad[f]; }
    es[i] = s; ed[i] = d;
}

// ---------------------------------------------------------------------------
// GAT aggregation (17 edges/target incl. self-loop) + bias + residual + LN1
// blockDim = 512, one block per target node.
// ---------------------------------------------------------------------------
__global__ void gat_agg_ln_kernel(const float* __restrict__ hp,
                                  const float* __restrict__ es, const float* __restrict__ ed,
                                  const int* __restrict__ srcIdx,
                                  const float* __restrict__ gbias,
                                  const float* __restrict__ hres,
                                  const float* __restrict__ g, const float* __restrict__ bb,
                                  float* __restrict__ uF, __bf16* __restrict__ uB) {
    const int t   = blockIdx.x;
    const int tid = threadIdx.x;
    const int hh  = tid >> 6;
    __shared__ int   ssrc[KNN_ + 1];
    __shared__ float s1[DM_], s2[DM_];
    if (tid < KNN_) ssrc[tid] = srcIdx[t * KNN_ + tid];
    if (tid == KNN_) ssrc[KNN_] = t;   // self loop
    __syncthreads();

    const float edt = ed[t * H_ + hh];
    float m = -1e30f;
#pragma unroll
    for (int j = 0; j < KNN_ + 1; ++j) {
        float e = es[ssrc[j] * H_ + hh] + edt;
        e = e > 0.f ? e : 0.2f * e;                    // LeakyReLU(0.2)
        m = fmaxf(m, e);
    }
    float z = 0.f, acc = 0.f;
#pragma unroll
    for (int j = 0; j < KNN_ + 1; ++j) {
        float e = es[ssrc[j] * H_ + hh] + edt;
        e = e > 0.f ? e : 0.2f * e;
        const float w = __expf(e - m);
        z += w;
        acc += w * hp[(size_t)ssrc[j] * DM_ + tid];    // tid == h*64+f
    }
    float v = acc / z + gbias[tid] + hres[(size_t)t * DM_ + tid];

    s1[tid] = v; s2[tid] = v * v;
    __syncthreads();
    for (int s = DM_ / 2; s > 0; s >>= 1) {
        if (tid < s) { s1[tid] += s1[tid + s]; s2[tid] += s2[tid + s]; }
        __syncthreads();
    }
    const float mu  = s1[0] * (1.f / DM_);
    const float var = s2[0] * (1.f / DM_) - mu * mu;
    const float o = (v - mu) * rsqrtf(var + 1e-5f) * g[tid] + bb[tid];
    uF[(size_t)t * DM_ + tid] = o;
    uB[(size_t)t * DM_ + tid] = f2bf(o);
}

// ---------------------------------------------------------------------------
// Global mean pool (nodes are contiguous per graph)
// ---------------------------------------------------------------------------
__global__ void pool_kernel(const float* __restrict__ h, float* __restrict__ pooled) {
    const int gph = blockIdx.x;
    const int d   = threadIdx.x;          // 512
    const float* __restrict__ base = h + (size_t)gph * NPG_ * DM_ + d;
    float s = 0.f;
    for (int i = 0; i < NPG_; ++i) s += base[(size_t)i * DM_];
    pooled[gph * DM_ + d] = s * (1.f / (float)NPG_);
}

// ---------------------------------------------------------------------------
// Classifier: [G,512] @ [512,16] + b
// ---------------------------------------------------------------------------
__global__ void cls_kernel(const float* __restrict__ pooled, const float* __restrict__ W,
                           const float* __restrict__ b, float* __restrict__ out) {
    const int tid = threadIdx.x;          // 256
    const int gph = tid >> 4, o = tid & 15;
    float s = b[o];
#pragma unroll 8
    for (int d = 0; d < DM_; ++d) s += pooled[gph * DM_ + d] * W[d * DOUT_ + o];
    out[tid] = s;
}

// ---------------------------------------------------------------------------
// Host-side orchestration
// ---------------------------------------------------------------------------
extern "C" void kernel_launch(void* const* d_in, const int* in_sizes, int n_in,
                              void* d_out, int out_size, void* d_ws, size_t ws_size,
                              hipStream_t stream) {
    (void)in_sizes; (void)n_in; (void)out_size; (void)ws_size;

    const float* x    = (const float*)d_in[0];
    const float* pos  = (const float*)d_in[1];
    // d_in[2] = batch (int32), unused: nodes are contiguous per graph.
    const float* enc_W1 = (const float*)d_in[3];
    const float* enc_b1 = (const float*)d_in[4];
    const float* enc_W2 = (const float*)d_in[5];
    const float* enc_b2 = (const float*)d_in[6];
    const float* enc_g  = (const float*)d_in[7];
    const float* enc_bt = (const float*)d_in[8];
    const float* gat_W  = (const float*)d_in[9];    // [L,512,512]
    const float* att_src= (const float*)d_in[10];   // [L,8,64]
    const float* att_dst= (const float*)d_in[11];
    const float* gat_b  = (const float*)d_in[12];   // [L,512]
    const float* ln1_g  = (const float*)d_in[13];
    const float* ln1_b  = (const float*)d_in[14];
    const float* ff_W1  = (const float*)d_in[15];   // [L,512,1024]
    const float* ff_b1  = (const float*)d_in[16];   // [L,1024]
    const float* ff_W2  = (const float*)d_in[17];   // [L,1024,512]
    const float* ff_b2  = (const float*)d_in[18];
    const float* ln2_g  = (const float*)d_in[19];
    const float* ln2_b  = (const float*)d_in[20];
    const float* fin_g  = (const float*)d_in[21];
    const float* fin_b  = (const float*)d_in[22];
    const float* cls_W  = (const float*)d_in[23];
    const float* cls_b  = (const float*)d_in[24];

    // ---- workspace carving ----
    char* ws = (char*)d_ws;
    size_t off = 0;
    auto carve = [&](size_t bytes) -> char* {
        char* p = ws + off;
        off += (bytes + 255) & ~(size_t)255;
        return p;
    };
    int*    srcIdx = (int*)   carve((size_t)NN_ * KNN_ * 4);
    float*  h_f32  = (float*) carve((size_t)NN_ * DM_ * 4);
    __bf16* h_bf   = (__bf16*)carve((size_t)NN_ * DM_ * 2);
    float*  tmp    = (float*) carve((size_t)NN_ * DM_ * 4);
    float*  u_f32  = (float*) carve((size_t)NN_ * DM_ * 4);
    __bf16* u_bf   = (__bf16*)carve((size_t)NN_ * DM_ * 2);
    __bf16* ffh_bf = (__bf16*)carve((size_t)NN_ * 2 * DM_ * 2); // also reused as h1_bf
    float*  es     = (float*) carve((size_t)NN_ * H_ * 4);
    float*  ed     = (float*) carve((size_t)NN_ * H_ * 4);
    float*  pooled = (float*) carve((size_t)G_ * DM_ * 4);
    __bf16* encW2T = (__bf16*)carve((size_t)256 * DM_ * 2);
    __bf16* gatWT  = (__bf16*)carve((size_t)L_ * DM_ * DM_ * 2);
    __bf16* ffW1T  = (__bf16*)carve((size_t)L_ * DM_ * 2 * DM_ * 2);
    __bf16* ffW2T  = (__bf16*)carve((size_t)L_ * 2 * DM_ * DM_ * 2);
    __bf16* h1_bf  = ffh_bf;   // transient, disjoint in time

    auto gemm = [&](const __bf16* A, const __bf16* BT, const float* bias,
                    float* Cf, __bf16* Cb, int M, int N, int Kd, int relu) {
        const int blocks = (M >> 6) * (N >> 7);   // 64x128 block tiles
        gemm_bf16_wmma<<<blocks, 256, 0, stream>>>(A, BT, bias, Cf, Cb, M, N, Kd, relu);
    };
    auto convt = [&](const float* W, __bf16* WT, int Kd, int N) {
        convt_kernel<<<(Kd * N + 255) / 256, 256, 0, stream>>>(W, WT, Kd, N);
    };

    // 1) KNN graph
    knn_kernel<<<NN_ / 256, 256, 0, stream>>>(pos, srcIdx);

    // 2) weight conversion (f32 -> bf16 transposed [N][K])
    convt(enc_W2, encW2T, 256, DM_);
    for (int l = 0; l < L_; ++l) {
        convt(gat_W + (size_t)l * DM_ * DM_,   gatWT + (size_t)l * DM_ * DM_,   DM_,     DM_);
        convt(ff_W1 + (size_t)l * DM_ * 2*DM_, ffW1T + (size_t)l * DM_ * 2*DM_, DM_,     2*DM_);
        convt(ff_W2 + (size_t)l * 2*DM_ * DM_, ffW2T + (size_t)l * 2*DM_ * DM_, 2 * DM_, DM_);
    }

    // 3) encoder
    enc1_kernel<<<NN_, 256, 0, stream>>>(x, pos, enc_W1, enc_b1, h1_bf);
    gemm(h1_bf, encW2T, enc_b2, tmp, nullptr, NN_, DM_, 256, 0);
    ln_kernel<<<NN_, DM_, 0, stream>>>(tmp, nullptr, enc_g, enc_bt, h_f32, h_bf);

    // 4) GAT layers
    for (int l = 0; l < L_; ++l) {
        // per-head linear projection (WMMA)
        gemm(h_bf, gatWT + (size_t)l * DM_ * DM_, nullptr, tmp, nullptr, NN_, DM_, DM_, 0);
        // attention scores
        att_score_kernel<<<NN_ * H_ / 256, 256, 0, stream>>>(
            tmp, att_src + (size_t)l * DM_, att_dst + (size_t)l * DM_, es, ed);
        // edge softmax + aggregation + bias + residual + LN1
        gat_agg_ln_kernel<<<NN_, DM_, 0, stream>>>(
            tmp, es, ed, srcIdx, gat_b + (size_t)l * DM_, h_f32,
            ln1_g + (size_t)l * DM_, ln1_b + (size_t)l * DM_, u_f32, u_bf);
        // feed-forward (WMMA x2)
        gemm(u_bf, ffW1T + (size_t)l * DM_ * 2*DM_, ff_b1 + (size_t)l * 2*DM_,
             nullptr, ffh_bf, NN_, 2 * DM_, DM_, 1);
        gemm(ffh_bf, ffW2T + (size_t)l * 2*DM_ * DM_, ff_b2 + (size_t)l * DM_,
             tmp, nullptr, NN_, DM_, 2 * DM_, 0);
        // residual + LN2 -> next h
        ln_kernel<<<NN_, DM_, 0, stream>>>(
            tmp, u_f32, ln2_g + (size_t)l * DM_, ln2_b + (size_t)l * DM_, h_f32, h_bf);
    }

    // 5) final LN -> pool -> classifier
    ln_kernel<<<NN_, DM_, 0, stream>>>(h_f32, nullptr, fin_g, fin_b, u_f32, nullptr);
    pool_kernel<<<G_, DM_, 0, stream>>>(u_f32, pooled);
    cls_kernel<<<1, 256, 0, stream>>>(pooled, cls_W, cls_b, (float*)d_out);
}
